// Mamba2Block_14620068676099
// MI455X (gfx1250) — matrix operand order repaired and checked by reading, not compiled, not run on hardware
//
#include <hip/hip_runtime.h>
#include <hip/hip_bf16.h>

typedef unsigned short u16;
typedef __bf16 bf16_t;
typedef bf16_t v16bf __attribute__((ext_vector_type(16)));
typedef float  v8f   __attribute__((ext_vector_type(8)));

union ABfrag { uint4 q[2]; v16bf v; };

// ---------- constants ----------
#define BB  4
#define LL  4096
#define DM  1024
#define EE  2048
#define HH  128
#define RR  64
#define KK  4
#define MM  (BB*LL)        // 16384
#define NP1 4096           // xp+z columns of proj we actually compute
#define NCH 64             // scan chunks per batch
#define CLH 64             // chunk length (NCH*CLH == LL)

// ---------- CDNA5 async global->LDS path (guarded; falls back to sync copy) ----
#if defined(__AMDGCN__) && __has_builtin(__builtin_amdgcn_global_load_async_to_lds_b128)
#define USE_ASYNC 1
#else
#define USE_ASYNC 0
#endif

typedef int v4i_vs __attribute__((vector_size(16)));
typedef __attribute__((address_space(1))) v4i_vs* gbl_v4i_p;
typedef __attribute__((address_space(3))) v4i_vs* lds_v4i_p;

__device__ __forceinline__ lds_v4i_p to_lds(const void* p) {
  return (lds_v4i_p)(unsigned)(size_t)p;      // generic LDS ptr: addr[31:0] = LDS offset
}
__device__ __forceinline__ gbl_v4i_p to_gbl(const void* p) {
  return (gbl_v4i_p)(size_t)p;
}
__device__ __forceinline__ void wait_async0() {
#if defined(__AMDGCN__)
#if __has_builtin(__builtin_amdgcn_s_wait_asynccnt)
  __builtin_amdgcn_s_wait_asynccnt(0);
#else
  asm volatile("s_wait_asynccnt 0x0" ::: "memory");
#endif
#endif
}

__device__ __forceinline__ u16 f2bf(float f) {
  unsigned u = __float_as_uint(f);
  unsigned r = (u + 0x7FFFu + ((u >> 16) & 1u)) >> 16;
  return (u16)r;
}
__device__ __forceinline__ float softplus_f(float x) {
  return (x > 20.f) ? x : log1pf(__expf(x));
}
__device__ __forceinline__ float sigmoid_f(float x) {
  return 1.f / (1.f + __expf(-x));
}

// ---------- elementwise converts ----------
__global__ void k_cvt_bf16(const float* __restrict__ src, u16* __restrict__ dst, size_t n) {
  size_t i = (size_t)blockIdx.x * blockDim.x + threadIdx.x;
  if (i < n) dst[i] = f2bf(src[i]);
}

// dst[n*rows + k] = bf16(src[k*cols + n])   (weight transpose to N-major)
__global__ void k_transpose_bf(const float* __restrict__ src, u16* __restrict__ dst,
                               int rows, int cols, int nsel) {
  size_t total = (size_t)rows * nsel;
  size_t i = (size_t)blockIdx.x * blockDim.x + threadIdx.x;
  if (i >= total) return;
  int n = (int)(i % nsel);
  int k = (int)(i / nsel);
  dst[(size_t)n * rows + k] = f2bf(src[(size_t)k * cols + n]);
}

// ---------- Bm0 = softplus( x . Win[:,4096] ) ----------
__global__ void k_bm0(const float* __restrict__ x, const float* __restrict__ Win,
                      float* __restrict__ bm0) {
  __shared__ float wcol[DM];
  int tid = threadIdx.x;
  for (int j = tid; j < DM; j += 256) wcol[j] = Win[(size_t)j * (3*EE) + 4096];
  __syncthreads();
  int w = tid >> 5, lane = tid & 31;
  size_t row = (size_t)blockIdx.x * 8 + w;
  const float* xr = x + row * DM;
  float s = 0.f;
  for (int k = lane; k < DM; k += 32) s += xr[k] * wcol[k];
  for (int m = 16; m >= 1; m >>= 1) s += __shfl_xor(s, m);
  if (lane == 0) bm0[row] = softplus_f(s);
}

// ---------- WMMA bf16 GEMM:  C(MxN,f32) = A(MxK,bf16 rowmajor) * Bt(NxK,bf16)^T ----------
// Async double-buffered LDS pipeline when GLOBAL_LOAD_ASYNC_TO_LDS is available.
template<int MT, int NT, int WR, int WC>
__global__ __launch_bounds__(256) void k_gemm_bf16(const u16* __restrict__ A,
                                                   const u16* __restrict__ Bt,
                                                   float* __restrict__ C,
                                                   int Mdim, int Ndim, int Kdim,
                                                   const float* __restrict__ scale) {
  constexpr int BM  = 16 * MT * WR;
  constexpr int BN  = 16 * NT * WC;
  constexpr int BK  = 32;
  constexpr int LDK = 40;                 // padded stride (16B align + bank spread)
  constexpr int ABUF = BM * LDK;
  constexpr int BBUF = BN * LDK;
#if USE_ASYNC
  __shared__ __align__(16) u16 sA[2 * ABUF];
  __shared__ __align__(16) u16 sB[2 * BBUF];
#else
  __shared__ __align__(16) u16 sA[ABUF];
  __shared__ __align__(16) u16 sB[BBUF];
#endif

  const int tid  = threadIdx.x;
  const int bmr  = blockIdx.y * BM;
  const int bnr  = blockIdx.x * BN;
  const int wid  = tid >> 5;
  const int lane = tid & 31;
  const int half = lane >> 4;
  const int ml   = lane & 15;
  const int wm   = (wid / WC) * (16 * MT);
  const int wn   = (wid % WC) * (16 * NT);

  v8f acc[MT][NT];
  for (int i = 0; i < MT; i++)
    for (int j = 0; j < NT; j++)
      for (int e = 0; e < 8; e++) acc[i][j][e] = 0.f;

  // ---- compute stage: fragment loads (contiguous k = 16*half + s in A and B) + WMMA ----
  auto compute = [&](int bi) {
    const u16* baseA = &sA[bi * ABUF];
    const u16* baseB = &sB[bi * BBUF];
    ABfrag a[MT];
#pragma unroll
    for (int i = 0; i < MT; i++) {
      const u16* p = &baseA[(wm + i * 16 + ml) * LDK];
      a[i].q[0] = *reinterpret_cast<const uint4*>(&p[16 * half]);       // K = 16*half+0..7
      a[i].q[1] = *reinterpret_cast<const uint4*>(&p[16 * half + 8]);   // K = 16*half+8..15
    }
#pragma unroll
    for (int j = 0; j < NT; j++) {
      ABfrag b;
      const u16* p = &baseB[(wn + j * 16 + ml) * LDK + 16 * half];      // K = 16*half+0..15
      b.q[0] = *reinterpret_cast<const uint4*>(&p[0]);
      b.q[1] = *reinterpret_cast<const uint4*>(&p[8]);
#pragma unroll
      for (int i = 0; i < MT; i++)
        acc[i][j] = __builtin_amdgcn_wmma_f32_16x16x32_bf16(
            false, a[i].v, false, b.v, (short)0, acc[i][j], false, false);
    }
  };

#if USE_ASYNC
  // ---- async DMA of one K-tile into buffer bi (16B per lane per instruction) ----
  // Structurally separate A-tile and B-tile instruction loops: all address math is
  // straight-line, no lane-dependent selects (avoids exec-mask codegen).
  constexpr int NA = BM / 64;             // async instrs per wave for A tile
  constexpr int NB = BN / 64;             // async instrs per wave for B tile
  auto issue_tile = [&](int bi, int k0) {
#pragma unroll
    for (int q = 0; q < NA; q++) {
      int c   = (wid * NA + q) * 32 + lane;    // 16B-chunk id in A tile
      int row = c >> 2;                        // 4 chunks per 32-elem row
      int kk  = (c & 3) * 8;
      __builtin_amdgcn_global_load_async_to_lds_b128(
          to_gbl(&A[(size_t)(bmr + row) * Kdim + k0 + kk]),
          to_lds(&sA[bi * ABUF + row * LDK + kk]), 0, 0);
    }
#pragma unroll
    for (int q = 0; q < NB; q++) {
      int c   = (wid * NB + q) * 32 + lane;    // 16B-chunk id in B tile
      int row = c >> 2;
      int kk  = (c & 3) * 8;
      __builtin_amdgcn_global_load_async_to_lds_b128(
          to_gbl(&Bt[(size_t)(bnr + row) * Kdim + k0 + kk]),
          to_lds(&sB[bi * BBUF + row * LDK + kk]), 0, 0);
    }
  };

  issue_tile(0, 0);
  int bi = 0;
  for (int k0 = 0; k0 < Kdim; k0 += BK) {
    wait_async0();                 // this wave's tile for buffer bi has landed in LDS
    __syncthreads();               // everyone's tile landed; prev buffer fully consumed
    if (k0 + BK < Kdim) issue_tile(bi ^ 1, k0 + BK);
    compute(bi);
    bi ^= 1;
  }
#else
  // ---- synchronous fallback: global->VGPR->LDS single-buffered ----
  constexpr int CA = (BM * BK) / (256 * 8);
  constexpr int CB = (BN * BK) / (256 * 8);
  for (int k0 = 0; k0 < Kdim; k0 += BK) {
    __syncthreads();
#pragma unroll
    for (int c = 0; c < CA; c++) {
      int chunk = tid + c * 256;
      int row = chunk >> 2;
      int kk  = (chunk & 3) * 8;
      *reinterpret_cast<uint4*>(&sA[row * LDK + kk]) =
          *reinterpret_cast<const uint4*>(&A[(size_t)(bmr + row) * Kdim + k0 + kk]);
    }
#pragma unroll
    for (int c = 0; c < CB; c++) {
      int chunk = tid + c * 256;
      int row = chunk >> 2;
      int kk  = (chunk & 3) * 8;
      *reinterpret_cast<uint4*>(&sB[row * LDK + kk]) =
          *reinterpret_cast<const uint4*>(&Bt[(size_t)(bnr + row) * Kdim + k0 + kk]);
    }
    __syncthreads();
    compute(0);
  }
#endif

  const float sc = scale ? *scale : 1.0f;
#pragma unroll
  for (int i = 0; i < MT; i++)
#pragma unroll
    for (int j = 0; j < NT; j++) {
      int r0 = bmr + wm + i * 16 + 8 * half;  // D: VGPR v -> row v + 8*half
      int c0 = bnr + wn + j * 16 + ml;        //          -> col lane%16
#pragma unroll
      for (int v = 0; v < 8; v++)
        C[(size_t)(r0 + v) * Ndim + c0] = acc[i][j][v] * sc;
    }
}

// ---------- causal depthwise conv (K=4) + SiLU ----------
__global__ void k_conv_silu(const float* __restrict__ proj, const float* __restrict__ cw,
                            const float* __restrict__ cb, float* __restrict__ xs,
                            u16* __restrict__ xs_bf) {
  size_t id = (size_t)blockIdx.x * blockDim.x + threadIdx.x;
  if (id >= (size_t)MM * EE) return;
  int e = (int)(id % EE);
  size_t row = id / EE;
  int t = (int)(row & (LL - 1));
  size_t brow = row - t;                // b*L
  float acc = cb[e];
#pragma unroll
  for (int k = 0; k < KK; k++) {
    int tt = t + k - (KK - 1);
    if (tt >= 0) acc += proj[(brow + tt) * NP1 + e] * cw[e * KK + k];
  }
  float s = acc * sigmoid_f(acc);
  xs[id] = s;
  xs_bf[id] = f2bf(s);
}

// ---------- dlt0 / bd ----------
__global__ void k_dlt_bd(const float* __restrict__ xdbl, const float* __restrict__ Wdt,
                         const float* __restrict__ bdt, const float* __restrict__ bm0,
                         const float* __restrict__ xs, float* __restrict__ dlt0,
                         float* __restrict__ bd) {
  __shared__ float wc[RR];
  int tid = threadIdx.x;
  if (tid < RR) wc[tid] = Wdt[(size_t)tid * EE];   // Wdt[:,0]
  __syncthreads();
  int w = tid >> 5, lane = tid & 31;
  size_t row = (size_t)blockIdx.x * 8 + w;
  const float* xr = xdbl + row * (RR + HH);
  float s = xr[lane] * wc[lane] + xr[lane + 32] * wc[lane + 32];
  for (int m = 16; m >= 1; m >>= 1) s += __shfl_xor(s, m);
  if (lane == 0) {
    float d = softplus_f(s + bdt[0]);
    d = fminf(fmaxf(d, 1e-6f), 100.f);
    dlt0[row] = d;
    bd[row] = d * bm0[row] * xs[row * EE];   // * xs[:,:,0]
  }
}

// ---------- chunked scan: phase 1 (local) ----------
__global__ void k_scan1(const float* __restrict__ dlt0, const float* __restrict__ bd,
                        const float* __restrict__ A_log, float* __restrict__ sloc,
                        float* __restrict__ ecum, float* __restrict__ sfin,
                        float* __restrict__ etot) {
  int h = threadIdx.x;                 // 0..127
  int bc = blockIdx.x;                 // b*NCH + c
  int b = bc >> 6, c = bc & (NCH - 1);
  float A0 = -__expf(A_log[h]);        // A[0][h]
  size_t i0 = ((size_t)b << 12) + (size_t)c * CLH;
  float s = 0.f, e = 1.f;
  for (int t = 0; t < CLH; t++) {
    size_t i = i0 + t;
    float a = __expf(dlt0[i] * A0);
    s = a * s + bd[i];
    e *= a;
    sloc[i * HH + h] = s;
    ecum[i * HH + h] = e;
  }
  size_t idx = (size_t)bc * HH + h;
  sfin[idx] = s;
  etot[idx] = e;
}

// ---------- chunked scan: phase 2 (combine chunk carries, per batch) ----------
__global__ void k_scan2(const float* __restrict__ sfin, const float* __restrict__ etot,
                        float* __restrict__ sin_) {
  int h = threadIdx.x;
  int b = blockIdx.x;
  float carry = 0.f;
  for (int c = 0; c < NCH; c++) {
    size_t idx = ((size_t)(b * NCH + c)) * HH + h;
    sin_[idx] = carry;
    carry = etot[idx] * carry + sfin[idx];
  }
}

// ---------- chunked scan: phase 3 (y_sc = sum_h C*(sloc + ecum*s_in)) ----------
__global__ void k_scan3(const float* __restrict__ xdbl, const float* __restrict__ sloc,
                        const float* __restrict__ ecum, const float* __restrict__ sin_,
                        float* __restrict__ ysc) {
  int tid = threadIdx.x;
  int w = tid >> 5, lane = tid & 31;
  size_t i = (size_t)blockIdx.x * 8 + w;
  int b = (int)(i >> 12);
  int c = ((int)i & (LL - 1)) >> 6;
  size_t sbase = ((size_t)(b * NCH + c)) * HH;
  const float* Cr = xdbl + i * (RR + HH) + RR;
  float s = 0.f;
#pragma unroll
  for (int j = 0; j < 4; j++) {
    int h = lane + j * 32;
    s += Cr[h] * (sloc[i * HH + h] + ecum[i * HH + h] * sin_[sbase + h]);
  }
  for (int m = 16; m >= 1; m >>= 1) s += __shfl_xor(s, m);
  if (lane == 0) ysc[i] = s;
}

// ---------- fused y = ysc + D*xs, LayerNorm, gate by sigmoid(z), -> bf16 ----------
__global__ void k_ln_gate(const float* __restrict__ ysc, const float* __restrict__ xs,
                          const float* __restrict__ proj, const float* __restrict__ Dv,
                          const float* __restrict__ g, const float* __restrict__ bta,
                          u16* __restrict__ yg_bf) {
  __shared__ float ybuf[EE];
  __shared__ float red[256], red2[256];
  int row = blockIdx.x, tid = threadIdx.x;
  float yv0 = ysc[row];
  float s = 0.f, s2 = 0.f;
  for (int e = tid; e < EE; e += 256) {
    float yv = yv0 + Dv[e] * xs[(size_t)row * EE + e];
    ybuf[e] = yv;
    s += yv; s2 += yv * yv;
  }
  red[tid] = s; red2[tid] = s2;
  __syncthreads();
  for (int st = 128; st >= 1; st >>= 1) {
    if (tid < st) { red[tid] += red[tid + st]; red2[tid] += red2[tid + st]; }
    __syncthreads();
  }
  float mu = red[0] * (1.f / EE);
  float var = red2[0] * (1.f / EE) - mu * mu;
  float rinv = rsqrtf(var + 1e-5f);
  for (int e = tid; e < EE; e += 256) {
    float yn = (ybuf[e] - mu) * rinv * g[e] + bta[e];
    float z = proj[(size_t)row * NP1 + EE + e];
    yg_bf[(size_t)row * EE + e] = f2bf(yn * sigmoid_f(z));
  }
}

// ---------- host ----------
extern "C" void kernel_launch(void* const* d_in, const int* in_sizes, int n_in,
                              void* d_out, int out_size, void* d_ws, size_t ws_size,
                              hipStream_t stream) {
  const float* x      = (const float*)d_in[0];
  const float* Win    = (const float*)d_in[1];
  const float* conv_w = (const float*)d_in[2];
  const float* conv_b = (const float*)d_in[3];
  const float* Wx     = (const float*)d_in[4];
  const float* Wdt    = (const float*)d_in[5];
  const float* bdt    = (const float*)d_in[6];
  const float* A_log  = (const float*)d_in[7];
  const float* Dv     = (const float*)d_in[8];
  const float* ln_g   = (const float*)d_in[9];
  const float* ln_b   = (const float*)d_in[10];
  const float* Wout   = (const float*)d_in[11];
  const float* rscale = (const float*)d_in[12];
  float* out = (float*)d_out;

  char* wp = (char*)d_ws;
  auto alloc = [&](size_t bytes) -> void* {
    void* p = (void*)wp;
    wp += (bytes + 255) & ~(size_t)255;
    return p;
  };
  u16*   x_bf   = (u16*)  alloc((size_t)MM * DM * 2);
  u16*   WinT   = (u16*)  alloc((size_t)NP1 * DM * 2);
  u16*   WxT    = (u16*)  alloc((size_t)(RR + HH) * EE * 2);
  u16*   WoutT  = (u16*)  alloc((size_t)DM * EE * 2);
  float* proj   = (float*)alloc((size_t)MM * NP1 * 4);
  float* bm0    = (float*)alloc((size_t)MM * 4);
  float* xs     = (float*)alloc((size_t)MM * EE * 4);
  u16*   xs_bf  = (u16*)  alloc((size_t)MM * EE * 2);
  float* xdbl   = (float*)alloc((size_t)MM * (RR + HH) * 4);
  float* dlt0   = (float*)alloc((size_t)MM * 4);
  float* bd     = (float*)alloc((size_t)MM * 4);
  float* ysc    = (float*)alloc((size_t)MM * 4);
  float* sloc   = (float*)alloc((size_t)MM * HH * 4);
  float* ecum   = (float*)alloc((size_t)MM * HH * 4);
  float* sfin   = (float*)alloc((size_t)BB * NCH * HH * 4);
  float* etot   = (float*)alloc((size_t)BB * NCH * HH * 4);
  float* sin_   = (float*)alloc((size_t)BB * NCH * HH * 4);
  u16*   yg_bf  = (u16*)  alloc((size_t)MM * EE * 2);

  // 1) converts / weight transposes
  {
    size_t n = (size_t)MM * DM;
    k_cvt_bf16<<<(unsigned)((n + 255) / 256), 256, 0, stream>>>(x, x_bf, n);
  }
  k_transpose_bf<<<(unsigned)(((size_t)DM * NP1 + 255) / 256), 256, 0, stream>>>(Win, WinT, DM, 3 * EE, NP1);
  k_transpose_bf<<<(unsigned)(((size_t)EE * (RR + HH) + 255) / 256), 256, 0, stream>>>(Wx, WxT, EE, RR + HH, RR + HH);
  k_transpose_bf<<<(unsigned)(((size_t)EE * DM + 255) / 256), 256, 0, stream>>>(Wout, WoutT, EE, DM, DM);

  // 2) Bm0 (proj column 4096 only — the only Bp element ever used)
  k_bm0<<<MM / 8, 256, 0, stream>>>(x, Win, bm0);

  // 3) GEMM1: proj[:, 0:4096] = x @ Win[:, 0:4096]   (xp | z)
  k_gemm_bf16<4, 4, 2, 4><<<dim3(NP1 / 256, MM / 128), 256, 0, stream>>>(
      x_bf, WinT, proj, MM, NP1, DM, nullptr);

  // 4) depthwise causal conv + SiLU
  k_conv_silu<<<(unsigned)(((size_t)MM * EE) / 256), 256, 0, stream>>>(proj, conv_w, conv_b, xs, xs_bf);

  // 5) GEMM2: x_dbl = xs @ Wx   (N=192)
  k_gemm_bf16<2, 2, 4, 2><<<dim3((RR + HH) / 64, MM / 128), 256, 0, stream>>>(
      xs_bf, WxT, xdbl, MM, RR + HH, EE, nullptr);

  // 6) dlt0 / bd scalars
  k_dlt_bd<<<MM / 8, 256, 0, stream>>>(xdbl, Wdt, bdt, bm0, xs, dlt0, bd);

  // 7) chunk-parallel scan
  k_scan1<<<BB * NCH, HH, 0, stream>>>(dlt0, bd, A_log, sloc, ecum, sfin, etot);
  k_scan2<<<BB, HH, 0, stream>>>(sfin, etot, sin_);
  k_scan3<<<MM / 8, 256, 0, stream>>>(xdbl, sloc, ecum, sin_, ysc);

  // 8) LN + gate -> yg (bf16)
  k_ln_gate<<<MM, 256, 0, stream>>>(ysc, xs, proj, Dv, ln_g, ln_b, yg_bf);

  // 9) GEMM3: out = yg @ Wout * res_scale
  k_gemm_bf16<4, 4, 2, 4><<<dim3(DM / 256, MM / 128), 256, 0, stream>>>(
      yg_bf, WoutT, out, MM, DM, EE, rscale);
}
